// PTC_39883066311166
// MI455X (gfx1250) — compile-verified, matrix operand order checked by reading, not run on hardware
//
#include <hip/hip_runtime.h>
#include <hip/hip_bf16.h>
#include <math.h>

// ---------------- problem constants (match reference) ----------------
#define BSZ 8
#define NPB 12500
#define NND (BSZ * NPB)       // 100000 fine nodes
#define DEG 16
#define GX 15
#define GY 15
#define NC (GX * GY)          // 225
#define N2C (BSZ * NC)        // 1800 coarse nodes
#define NUM_OUT 101
#define EPS_BN 1e-5f
#define ENC_NEGINF 0x007FFFFFu   // order-preserving encoding of -inf

typedef __attribute__((ext_vector_type(16))) _Float16 v16h;
typedef __attribute__((ext_vector_type(8)))  _Float16 v8h;
typedef __attribute__((ext_vector_type(8)))  float    v8f;

// ---------------- helpers ----------------
__device__ __forceinline__ unsigned encf(float f) {
    unsigned u = __float_as_uint(f);
    return (u & 0x80000000u) ? ~u : (u | 0x80000000u);
}
__device__ __forceinline__ float decf(unsigned u) {
    unsigned b = (u & 0x80000000u) ? (u & 0x7FFFFFFFu) : ~u;
    return __uint_as_float(b);
}
__device__ __forceinline__ float elu1(float x) {
    return x > 0.f ? x : (__expf(x) - 1.f);
}

// ---------------- operand packing for WMMA ----------------
// Ah[Mpad, Kpad] = zero-padded f16 copy of A[M, K]
__global__ void pack_a_f16_kernel(const float* __restrict__ A,
                                  _Float16* __restrict__ Ah,
                                  int M, int K, int Kpad, int total) {
    int idx = blockIdx.x * blockDim.x + threadIdx.x;
    if (idx >= total) return;
    int r = idx / Kpad, k = idx % Kpad;
    float v = 0.f;
    if (r < M && k < K) v = A[(size_t)r * K + k];
    Ah[idx] = (_Float16)v;
}
// Wt[Npad, Kpad] = zero-padded f16 TRANSPOSED copy of W[K, Ncol]
__global__ void pack_wt_f16_kernel(const float* __restrict__ W,
                                   _Float16* __restrict__ Wt,
                                   int K, int Ncol, int Kpad, int total) {
    int idx = blockIdx.x * blockDim.x + threadIdx.x;
    if (idx >= total) return;
    int n = idx / Kpad, k = idx % Kpad;
    float v = 0.f;
    if (n < Ncol && k < K) v = W[(size_t)k * Ncol + n];
    Wt[idx] = (_Float16)v;
}

// ---------------- WMMA GEMM on packed operands.
// C[M,Ncol] (f32) = Ah[Mpad,Kpad] * Wt[Npad,Kpad]^T via v_wmma_f32_16x16x32_f16.
// One wave per 16x16 C tile; per K-chunk: 4x global_load_b128 + 1x wmma,
// zero branches in the loop (operands pre-padded). ----------------
__global__ void gemm_wmma_kernel(const _Float16* __restrict__ Ah,
                                 const _Float16* __restrict__ Wt,
                                 float* __restrict__ C,
                                 int M, int Kpad, int Ncol) {
    int tilesN = (Ncol + 15) >> 4;
    int tm = blockIdx.x / tilesN;
    int tn = blockIdx.x % tilesN;
    int lane = threadIdx.x & 31;
    int half_sel = lane >> 4;      // 0: lanes 0-15, 1: lanes 16-31
    int lidx = lane & 15;

    // A: 16-bit 16x32 layout (ISA 7.12.2): lane holds row M=lidx,
    //    K = {0..7,16..23} (+8 for upper half-wave). Contiguous 8-half runs.
    const _Float16* ap = Ah + (size_t)(tm * 16 + lidx) * Kpad + half_sel * 8;
    // B mirrored: lane holds column N=lidx, same K pattern (Wt is N-major).
    const _Float16* bp = Wt + (size_t)(tn * 16 + lidx) * Kpad + half_sel * 8;

    v8f acc = {};
    for (int k0 = 0; k0 < Kpad; k0 += 32) {
        v8h a0 = *(const v8h*)(ap + k0);
        v8h a1 = *(const v8h*)(ap + k0 + 16);
        v8h b0 = *(const v8h*)(bp + k0);
        v8h b1 = *(const v8h*)(bp + k0 + 16);
        v16h a = __builtin_shufflevector(a0, a1, 0, 1, 2, 3, 4, 5, 6, 7,
                                         8, 9, 10, 11, 12, 13, 14, 15);
        v16h b = __builtin_shufflevector(b0, b1, 0, 1, 2, 3, 4, 5, 6, 7,
                                         8, 9, 10, 11, 12, 13, 14, 15);
        acc = __builtin_amdgcn_wmma_f32_16x16x32_f16(
            /*neg_a=*/false, a, /*neg_b=*/false, b,
            /*c_mod=*/(short)0, acc, /*reuse_a=*/false, /*reuse_b=*/false);
    }
    // C/D layout: lanes 0-15 -> M = v, lanes 16-31 -> M = v+8; N = lane&15
    int ocol = tn * 16 + lidx;
#pragma unroll
    for (int v = 0; v < 8; ++v) {
        int orow = tm * 16 + v + half_sel * 8;
        if (orow < M && ocol < Ncol) C[(size_t)orow * Ncol + ocol] = acc[v];
    }
}

// ---------------- PointTransformerConv attention aggregation.
// One thread per (dst node, channel). K incoming edges + self loop,
// all kept in registers (fully unrolled). ----------------
template <int K, bool COARSE>
__global__ void pt_conv_kernel(const float* __restrict__ h,
                               const float* __restrict__ asc,
                               const float* __restrict__ adc_,
                               const float* __restrict__ pos,
                               const int* __restrict__ esrc,
                               const float* __restrict__ w_pos,
                               const float* __restrict__ b_pos,
                               float* __restrict__ out,
                               int n, int cch) {
    int idx = blockIdx.x * blockDim.x + threadIdx.x;
    if (idx >= n * cch) return;
    int i = idx / cch, c = idx % cch;

    float px = pos[i * 3 + 0], py = pos[i * 3 + 1], pz = pos[i * 3 + 2];
    float adc = adc_[idx];
    float wp0 = w_pos[c], wp1 = w_pos[cch + c], wp2 = w_pos[2 * cch + c];
    float bp = b_pos[c];

    float alpha[K + 1], hd[K + 1];
#pragma unroll
    for (int k = 0; k < K; ++k) {
        int src;
        if (COARSE) {
            int b = i / NC, ii = i % NC;
            src = esrc[(b * 8 + k) * NC + ii];   // edge layout: batch,offset,node
        } else {
            src = esrc[i * K + k];               // dst = repeat(arange(N), DEG)
        }
        float dx = px - pos[src * 3 + 0];
        float dy = py - pos[src * 3 + 1];
        float dz = pz - pos[src * 3 + 2];
        float delta = dx * wp0 + dy * wp1 + dz * wp2 + bp;
        alpha[k] = adc - asc[(size_t)src * cch + c] + delta;
        hd[k]    = h[(size_t)src * cch + c] + delta;
    }
    // self loop (pos delta = 0)
    alpha[K] = adc - asc[idx] + bp;
    hd[K]    = h[idx] + bp;

    float amax = alpha[0];
#pragma unroll
    for (int k = 1; k <= K; ++k) amax = fmaxf(amax, alpha[k]);
    float denom = 0.f, num = 0.f;
#pragma unroll
    for (int k = 0; k <= K; ++k) {
        float e = __expf(alpha[k] - amax);
        denom += e;
        num += e * hd[k];
    }
    out[idx] = num / (denom + 1e-16f);
}

// ---------------- ELU + BN statistics: block per channel ----------------
__global__ void elu_bn_stats_kernel(const float* __restrict__ v,
                                    float* __restrict__ stats,
                                    int n, int cch) {
    __shared__ float sh0[256], sh1[256];
    int c = blockIdx.x;
    float s = 0.f, s2 = 0.f;
    for (int i = threadIdx.x; i < n; i += blockDim.x) {
        float xv = elu1(v[(size_t)i * cch + c]);
        s += xv;
        s2 += xv * xv;
    }
    sh0[threadIdx.x] = s;
    sh1[threadIdx.x] = s2;
    __syncthreads();
    for (int st = 128; st > 0; st >>= 1) {
        if ((int)threadIdx.x < st) {
            sh0[threadIdx.x] += sh0[threadIdx.x + st];
            sh1[threadIdx.x] += sh1[threadIdx.x + st];
        }
        __syncthreads();
    }
    if (threadIdx.x == 0) {
        stats[c] = sh0[0];
        stats[cch + c] = sh1[0];
    }
}

// ---------------- ELU + BN apply (+ optional residual) ----------------
__global__ void elu_bn_apply_kernel(const float* __restrict__ v,
                                    const float* __restrict__ stats,
                                    const float* __restrict__ g,
                                    const float* __restrict__ be,
                                    const float* __restrict__ res,
                                    float* __restrict__ out,
                                    int n, int cch) {
    int idx = blockIdx.x * blockDim.x + threadIdx.x;
    if (idx >= n * cch) return;
    int c = idx % cch;
    float inv_n = 1.f / (float)n;
    float mu = stats[c] * inv_n;
    float var = stats[cch + c] * inv_n - mu * mu;
    float xv = elu1(v[idx]);
    float y = g[c] * (xv - mu) * rsqrtf(var + EPS_BN) + be[c];
    if (res) y += res[idx];
    out[idx] = y;
}

// ---------------- fill kernels ----------------
__global__ void fill_u32_kernel(unsigned* p, unsigned v, int n) {
    int i = blockIdx.x * blockDim.x + threadIdx.x;
    if (i < n) p[i] = v;
}
__global__ void fill_f32_kernel(float* p, float v, int n) {
    int i = blockIdx.x * blockDim.x + threadIdx.x;
    if (i < n) p[i] = v;
}

// ---------------- pool3: voxel 16x12 grid max-pool + pos mean ----------------
__global__ void pool3_scatter_kernel(const float* __restrict__ x,
                                     const float* __restrict__ pos,
                                     const int* __restrict__ batch,
                                     unsigned* __restrict__ enc,
                                     float* __restrict__ possum,
                                     float* __restrict__ cnt) {
    int idx = blockIdx.x * blockDim.x + threadIdx.x;
    if (idx >= NND * 32) return;
    int i = idx >> 5, c = idx & 31;
    float px = pos[i * 3 + 0], py = pos[i * 3 + 1];
    int cx = min(max((int)floorf(px / 16.f), 0), GX - 1);
    int cy = min(max((int)floorf(py / 12.f), 0), GY - 1);
    int cl = batch[i] * NC + cy * GX + cx;
    atomicMax(&enc[cl * 32 + c], encf(x[idx]));
    if (c == 0) {
        atomicAdd(&cnt[cl], 1.f);
        atomicAdd(&possum[cl * 3 + 0], px);
        atomicAdd(&possum[cl * 3 + 1], py);
        atomicAdd(&possum[cl * 3 + 2], pos[i * 3 + 2]);
    }
}
__global__ void pool3_finalize_kernel(const unsigned* __restrict__ enc,
                                      float* __restrict__ x2,
                                      float* __restrict__ pos2,
                                      const float* __restrict__ cnt) {
    int idx = blockIdx.x * blockDim.x + threadIdx.x;
    if (idx >= N2C * 32) return;
    unsigned e = enc[idx];
    x2[idx] = (e == ENC_NEGINF) ? 0.f : decf(e);
    if ((idx & 31) == 0) {
        int cl = idx >> 5;
        float cn = fmaxf(cnt[cl], 1.f);
        pos2[cl * 3 + 0] /= cn;
        pos2[cl * 3 + 1] /= cn;
        pos2[cl * 3 + 2] /= cn;
    }
}

// ---------------- pool5: 4x4 grid max-pool over coarse nodes ----------------
__global__ void pool5_scatter_kernel(const float* __restrict__ x5,
                                     const float* __restrict__ pos2,
                                     unsigned* __restrict__ enc) {
    int idx = blockIdx.x * blockDim.x + threadIdx.x;
    if (idx >= N2C * 128) return;
    int j = idx >> 7, c = idx & 127;
    int b = j / NC;
    int cx = min(max((int)floorf(pos2[j * 3 + 0] / 60.f), 0), 3);
    int cy = min(max((int)floorf(pos2[j * 3 + 1] / 45.f), 0), 3);
    int cl = b * 16 + cy * 4 + cx;
    atomicMax(&enc[cl * 128 + c], encf(x5[idx]));
}
__global__ void pool5_finalize_kernel(const unsigned* __restrict__ enc,
                                      float* __restrict__ pooled) {
    int idx = blockIdx.x * blockDim.x + threadIdx.x;
    if (idx >= BSZ * 16 * 128) return;
    unsigned e = enc[idx];
    pooled[idx] = (e == ENC_NEGINF) ? 0.f : decf(e);
}

// ---------------- host-side orchestration ----------------
static inline int cdiv(int a, int b) { return (a + b - 1) / b; }

// Pack A once (shared by the 3 GEMMs of a layer), then per-W pack + GEMM.
static void pack_a(const float* A, _Float16* Ah, int M, int K,
                   int& Mpad, int& Kpad, hipStream_t s) {
    Mpad = cdiv(M, 16) * 16;
    Kpad = cdiv(K, 32) * 32;
    int total = Mpad * Kpad;
    pack_a_f16_kernel<<<cdiv(total, 256), 256, 0, s>>>(A, Ah, M, K, Kpad, total);
}
static void gemm_packed(const _Float16* Ah, const float* W, _Float16* Wt,
                        float* C, int M, int K, int Ncol,
                        int Mpad, int Kpad, hipStream_t s) {
    int Npad = cdiv(Ncol, 16) * 16;
    int totalW = Npad * Kpad;
    pack_wt_f16_kernel<<<cdiv(totalW, 256), 256, 0, s>>>(W, Wt, K, Ncol, Kpad, totalW);
    gemm_wmma_kernel<<<(Mpad / 16) * (Npad / 16), 32, 0, s>>>(Ah, Wt, C, M, Kpad, Ncol);
}

extern "C" void kernel_launch(void* const* d_in, const int* in_sizes, int n_in,
                              void* d_out, int out_size, void* d_ws, size_t ws_size,
                              hipStream_t stream) {
    (void)in_sizes; (void)n_in; (void)out_size; (void)ws_size;
    const float* x     = (const float*)d_in[0];
    const float* pos   = (const float*)d_in[1];
    const int*   batch = (const int*)d_in[2];
    const int*   e1    = (const int*)d_in[3];   // row0 = src (dst implied)
    const int*   e2    = (const int*)d_in[4];   // row0 = src (structured)
    auto WPp = [&](int l, int j) { return (const float*)d_in[5 + l * 7 + j]; };
    // j: 0=w_lin 1=w_src 2=w_dst 3=w_pos 4=b_pos 5=g 6=be
    const float* w_fc = (const float*)d_in[40];
    float* out = (float*)d_out;

    // -------- workspace layout (16B-aligned chunks) --------
    float* ws = (float*)d_ws;
    size_t o = 0;
    auto alloc = [&](size_t n) {
        float* p = ws + o;
        o += (n + 7) & ~(size_t)7;   // keep 32B alignment between chunks
        return p;
    };
    float* Xa  = alloc((size_t)NND * 32);
    float* Xb  = alloc((size_t)NND * 32);
    float* H   = alloc((size_t)NND * 32);
    float* AS  = alloc((size_t)NND * 32);
    float* AD  = alloc((size_t)NND * 32);
    float* CV  = alloc((size_t)NND * 32);
    float* STATS = alloc(256);
    float* X2C = alloc((size_t)N2C * 32);
    float* POS2 = alloc((size_t)N2C * 3);
    float* CNT  = alloc(N2C);
    unsigned* ENC3 = (unsigned*)alloc((size_t)N2C * 32);
    float* H2  = alloc((size_t)N2C * 128);
    float* AS2 = alloc((size_t)N2C * 128);
    float* AD2 = alloc((size_t)N2C * 128);
    float* CV2 = alloc((size_t)N2C * 128);
    float* X4  = alloc((size_t)N2C * 32);
    float* X5  = alloc((size_t)N2C * 128);
    unsigned* ENC5 = (unsigned*)alloc(BSZ * 16 * 128);
    float* POOLED  = alloc(BSZ * 16 * 128);
    _Float16* AH = (_Float16*)alloc((size_t)NND * 32 / 2 + 16);  // 3.2M halves
    _Float16* WT = (_Float16*)alloc(131072);                      // 262144 halves

    const int TB = 256;
    auto gridN = [&](int n) { return (n + TB - 1) / TB; };
    int Mpad, Kpad;

    // ======== layer 1 (fine, cin=1 -> cout=8) ========
    pack_a(x, AH, NND, 1, Mpad, Kpad, stream);
    gemm_packed(AH, WPp(0, 0), WT, H,  NND, 1, 8, Mpad, Kpad, stream);
    gemm_packed(AH, WPp(0, 1), WT, AS, NND, 1, 8, Mpad, Kpad, stream);
    gemm_packed(AH, WPp(0, 2), WT, AD, NND, 1, 8, Mpad, Kpad, stream);
    pt_conv_kernel<DEG, false><<<gridN(NND * 8), TB, 0, stream>>>(
        H, AS, AD, pos, e1, WPp(0, 3), WPp(0, 4), CV, NND, 8);
    elu_bn_stats_kernel<<<8, 256, 0, stream>>>(CV, STATS, NND, 8);
    elu_bn_apply_kernel<<<gridN(NND * 8), TB, 0, stream>>>(
        CV, STATS, WPp(0, 5), WPp(0, 6), nullptr, Xa, NND, 8);   // X1 = sc

    // ======== layer 2 (fine, 8 -> 8, residual X1) ========
    pack_a(Xa, AH, NND, 8, Mpad, Kpad, stream);
    gemm_packed(AH, WPp(1, 0), WT, H,  NND, 8, 8, Mpad, Kpad, stream);
    gemm_packed(AH, WPp(1, 1), WT, AS, NND, 8, 8, Mpad, Kpad, stream);
    gemm_packed(AH, WPp(1, 2), WT, AD, NND, 8, 8, Mpad, Kpad, stream);
    pt_conv_kernel<DEG, false><<<gridN(NND * 8), TB, 0, stream>>>(
        H, AS, AD, pos, e1, WPp(1, 3), WPp(1, 4), CV, NND, 8);
    elu_bn_stats_kernel<<<8, 256, 0, stream>>>(CV, STATS, NND, 8);
    elu_bn_apply_kernel<<<gridN(NND * 8), TB, 0, stream>>>(
        CV, STATS, WPp(1, 5), WPp(1, 6), Xa, Xb, NND, 8);        // X2

    // ======== layer 3 (fine, 8 -> 32) ========
    pack_a(Xb, AH, NND, 8, Mpad, Kpad, stream);
    gemm_packed(AH, WPp(2, 0), WT, H,  NND, 8, 32, Mpad, Kpad, stream);
    gemm_packed(AH, WPp(2, 1), WT, AS, NND, 8, 32, Mpad, Kpad, stream);
    gemm_packed(AH, WPp(2, 2), WT, AD, NND, 8, 32, Mpad, Kpad, stream);
    pt_conv_kernel<DEG, false><<<gridN(NND * 32), TB, 0, stream>>>(
        H, AS, AD, pos, e1, WPp(2, 3), WPp(2, 4), CV, NND, 32);
    elu_bn_stats_kernel<<<32, 256, 0, stream>>>(CV, STATS, NND, 32);
    elu_bn_apply_kernel<<<gridN(NND * 32), TB, 0, stream>>>(
        CV, STATS, WPp(2, 5), WPp(2, 6), nullptr, Xa, NND, 32);  // X3

    // ======== pool3 ========
    fill_u32_kernel<<<gridN(N2C * 32), TB, 0, stream>>>(ENC3, ENC_NEGINF, N2C * 32);
    fill_f32_kernel<<<gridN(N2C * 3), TB, 0, stream>>>(POS2, 0.f, N2C * 3);
    fill_f32_kernel<<<gridN(N2C), TB, 0, stream>>>(CNT, 0.f, N2C);
    pool3_scatter_kernel<<<gridN(NND * 32), TB, 0, stream>>>(Xa, pos, batch, ENC3, POS2, CNT);
    pool3_finalize_kernel<<<gridN(N2C * 32), TB, 0, stream>>>(ENC3, X2C, POS2, CNT);

    // ======== layer 4 (coarse, 32 -> 32, residual X2C) ========
    pack_a(X2C, AH, N2C, 32, Mpad, Kpad, stream);
    gemm_packed(AH, WPp(3, 0), WT, H2,  N2C, 32, 32, Mpad, Kpad, stream);
    gemm_packed(AH, WPp(3, 1), WT, AS2, N2C, 32, 32, Mpad, Kpad, stream);
    gemm_packed(AH, WPp(3, 2), WT, AD2, N2C, 32, 32, Mpad, Kpad, stream);
    pt_conv_kernel<8, true><<<gridN(N2C * 32), TB, 0, stream>>>(
        H2, AS2, AD2, POS2, e2, WPp(3, 3), WPp(3, 4), CV2, N2C, 32);
    elu_bn_stats_kernel<<<32, 256, 0, stream>>>(CV2, STATS, N2C, 32);
    elu_bn_apply_kernel<<<gridN(N2C * 32), TB, 0, stream>>>(
        CV2, STATS, WPp(3, 5), WPp(3, 6), X2C, X4, N2C, 32);

    // ======== layer 5 (coarse, 32 -> 128) ========
    pack_a(X4, AH, N2C, 32, Mpad, Kpad, stream);
    gemm_packed(AH, WPp(4, 0), WT, H2,  N2C, 32, 128, Mpad, Kpad, stream);
    gemm_packed(AH, WPp(4, 1), WT, AS2, N2C, 32, 128, Mpad, Kpad, stream);
    gemm_packed(AH, WPp(4, 2), WT, AD2, N2C, 32, 128, Mpad, Kpad, stream);
    pt_conv_kernel<8, true><<<gridN(N2C * 128), TB, 0, stream>>>(
        H2, AS2, AD2, POS2, e2, WPp(4, 3), WPp(4, 4), CV2, N2C, 128);
    elu_bn_stats_kernel<<<128, 256, 0, stream>>>(CV2, STATS, N2C, 128);
    elu_bn_apply_kernel<<<gridN(N2C * 128), TB, 0, stream>>>(
        CV2, STATS, WPp(4, 5), WPp(4, 6), nullptr, X5, N2C, 128);

    // ======== pool5 + FC ========
    fill_u32_kernel<<<gridN(BSZ * 16 * 128), TB, 0, stream>>>(ENC5, ENC_NEGINF, BSZ * 16 * 128);
    pool5_scatter_kernel<<<gridN(N2C * 128), TB, 0, stream>>>(X5, POS2, ENC5);
    pool5_finalize_kernel<<<gridN(BSZ * 16 * 128), TB, 0, stream>>>(ENC5, POOLED);

    pack_a(POOLED, AH, BSZ, 16 * 128, Mpad, Kpad, stream);
    gemm_packed(AH, w_fc, WT, out, BSZ, 16 * 128, NUM_OUT, Mpad, Kpad, stream);
}
